// PointNet2Critic_32461362823176
// MI455X (gfx1250) — matrix operand order repaired and checked by reading, not compiled, not run on hardware
//
#include <hip/hip_runtime.h>
#include <stdint.h>

typedef _Float16 half_t;
typedef __attribute__((ext_vector_type(16))) _Float16 v16h;
typedef __attribute__((ext_vector_type(8)))  _Float16 v8h;
typedef __attribute__((ext_vector_type(8)))  float    v8f;

union ABfrag { v16h v; v8h h[2]; };

// ---------------------------------------------------------------------------
// ply2data: env (B,3,32,32,16) -> x (B,N,9) f16, coords0 (N,3) f32 (batch-shared)
// ---------------------------------------------------------------------------
__global__ void ply2data_kernel(const float* __restrict__ env,
                                half_t* __restrict__ x, float* __restrict__ coords0) {
  int i = blockIdx.x * blockDim.x + threadIdx.x;
  if (i >= 4 * 16384) return;
  int b = i >> 14;
  int n = i & 16383;
  int w = n >> 9;            // n = (w*32 + d)*16 + h
  int d = (n >> 4) & 31;
  int h = n & 15;
  const float VL = 0.2f;
  float gx = w * VL, gy = d * VL, gz = h * VL;
  float dx = gx - 3.1f, dy = gy - 3.1f, dz = gz - 1.5f;   // grid - mean
  float nx = gx / 6.2f, ny = gy / 6.2f, nz = gz / 3.0f;   // (grid-min)/(max-min)
  size_t eb = (size_t)b * 3 * 16384;
  float occ    = env[eb + 0 * 16384 + n];
  float permit = env[eb + 1 * 16384 + n];
  float vis    = env[eb + 2 * 16384 + n];
  float cr = 0.f, cg = 0.f, cb = 0.f;
  if (occ != 0.f)    { cr = 128.f; cg = 128.f; cb = 128.f; }
  if (permit != 0.f) { cr = 255.f; cg = 255.f; cb = 0.f; }
  if (vis > 0.f)     { cr = 0.f;   cg = vis * 255.f; cb = 0.f; }
  cr *= (1.f / 255.f); cg *= (1.f / 255.f); cb *= (1.f / 255.f);
  half_t* row = x + (size_t)i * 9;
  row[0] = (half_t)dx; row[1] = (half_t)dy; row[2] = (half_t)dz;
  row[3] = (half_t)nx; row[4] = (half_t)ny; row[5] = (half_t)nz;
  row[6] = (half_t)cr; row[7] = (half_t)cg; row[8] = (half_t)cb;
  if (b == 0) {
    coords0[n * 3 + 0] = dx; coords0[n * 3 + 1] = dy; coords0[n * 3 + 2] = dz;
  }
}

// ---------------------------------------------------------------------------
// Farthest point sampling: one block, LDS argmax tree reduction each step.
// ---------------------------------------------------------------------------
__global__ void fps_kernel(const float* __restrict__ coords, int n, int npoint,
                           float* __restrict__ newxyz, float* __restrict__ dist) {
  int t = threadIdx.x;
  __shared__ float sval[256];
  __shared__ int   sidx[256];
  __shared__ int   sfar;
  for (int j = t; j < n; j += 256) dist[j] = 3.4e38f;
  if (t == 0) sfar = 0;
  __syncthreads();
  for (int s = 0; s < npoint; ++s) {
    int far = sfar;
    float cx = coords[far * 3 + 0];
    float cy = coords[far * 3 + 1];
    float cz = coords[far * 3 + 2];
    if (t == 0) { newxyz[s * 3 + 0] = cx; newxyz[s * 3 + 1] = cy; newxyz[s * 3 + 2] = cz; }
    float best = -1.f; int bi = 0;
    for (int j = t; j < n; j += 256) {
      float ax = coords[j * 3 + 0] - cx;
      float ay = coords[j * 3 + 1] - cy;
      float az = coords[j * 3 + 2] - cz;
      float dd = ax * ax + ay * ay + az * az;
      float nd = fminf(dist[j], dd);
      dist[j] = nd;
      if (nd > best) { best = nd; bi = j; }
    }
    sval[t] = best; sidx[t] = bi;
    __syncthreads();
    for (int o = 128; o > 0; o >>= 1) {
      if (t < o && sval[t + o] > sval[t]) { sval[t] = sval[t + o]; sidx[t] = sidx[t + o]; }
      __syncthreads();
    }
    if (t == 0) sfar = sidx[0];
    __syncthreads();
  }
}

// ---------------------------------------------------------------------------
// Ball query: 128 centers per block; point chunks staged into LDS via CDNA5
// async-to-LDS (ASYNCcnt + s_wait_asynccnt), reused by all centers in block.
// ---------------------------------------------------------------------------
#define QB_BLK 128
__global__ void query_ball_kernel(const float* __restrict__ coords, int n,
                                  const float* __restrict__ newxyz, int npoint,
                                  float r2, int* __restrict__ ballidx) {
  __shared__ float sch[QB_BLK * 3];
  int t = threadIdx.x;
  int s = blockIdx.x * QB_BLK + t;
  bool ok = s < npoint;
  float cx = 0.f, cy = 0.f, cz = 0.f;
  if (ok) { cx = newxyz[s * 3 + 0]; cy = newxyz[s * 3 + 1]; cz = newxyz[s * 3 + 2]; }
  int cnt = 0, first = -1;
  for (int base = 0; base < n; base += QB_BLK) {
    int m = n - base; if (m > QB_BLK) m = QB_BLK;
    int nel = m * 3;
    unsigned long long gb = (unsigned long long)(uintptr_t)(coords + (size_t)base * 3);
#pragma unroll
    for (int i = 0; i < 3; ++i) {
      int e = i * QB_BLK + t;
      if (e < nel) {
        unsigned lds = (unsigned)(uintptr_t)(&sch[e]);   // low 32 bits == LDS byte addr
        unsigned voff = (unsigned)(e * 4);
        asm volatile("global_load_async_to_lds_b32 %0, %1, %2"
                     :: "v"(lds), "v"(voff), "s"(gb) : "memory");
      }
    }
    asm volatile("s_wait_asynccnt 0x0" ::: "memory");
    __syncthreads();
    if (ok) {
      for (int j = 0; j < m && cnt < 32; ++j) {
        float ax = sch[j * 3 + 0] - cx;
        float ay = sch[j * 3 + 1] - cy;
        float az = sch[j * 3 + 2] - cz;
        float dd = ax * ax + ay * ay + az * az;
        if (dd <= r2) {
          int gi = base + j;
          if (first < 0) first = gi;
          ballidx[s * 32 + cnt] = gi;
          ++cnt;
        }
      }
    }
    __syncthreads();
  }
  if (ok) {
    if (first < 0) first = 0;
    for (; cnt < 32; ++cnt) ballidx[s * 32 + cnt] = first;
  }
}

// ---------------------------------------------------------------------------
// Build grouped GEMM input rows: [gxyz(3) | feats(C) | zero-pad] width Kc.
// ---------------------------------------------------------------------------
__global__ void group_kernel(const half_t* __restrict__ feats, int C,
                             const float* __restrict__ coords,
                             const float* __restrict__ newxyz,
                             const int* __restrict__ ballidx,
                             half_t* __restrict__ A, int npoint, int Kc) {
  int r = blockIdx.x * blockDim.x + threadIdx.x;   // r = s*32 + j
  if (r >= npoint * 32) return;
  int s = r >> 5;
  int p = ballidx[r];
  half_t* row = A + (size_t)r * Kc;
  row[0] = (half_t)(coords[p * 3 + 0] - newxyz[s * 3 + 0]);
  row[1] = (half_t)(coords[p * 3 + 1] - newxyz[s * 3 + 1]);
  row[2] = (half_t)(coords[p * 3 + 2] - newxyz[s * 3 + 2]);
  const half_t* f = feats + (size_t)p * C;
  for (int c = 0; c < C; ++c) row[3 + c] = f[c];
  for (int c = 3 + C; c < Kc; ++c) row[c] = (half_t)0.f;
}

// ---------------------------------------------------------------------------
// Weight/pad convert: f32 (N,Kw) row-major -> f16 (N,Kc) zero-padded.
// ---------------------------------------------------------------------------
__global__ void convert_w_kernel(const float* __restrict__ W, half_t* __restrict__ Wh,
                                 int N, int Kw, int Kc) {
  int i = blockIdx.x * blockDim.x + threadIdx.x;
  if (i >= N * Kc) return;
  int n = i / Kc, k = i % Kc;
  Wh[i] = (half_t)((k < Kw) ? W[(size_t)n * Kw + k] : 0.f);
}

// ---------------------------------------------------------------------------
// WMMA GEMM: C(M,N) = act(A(M,Kc)f16 @ Wh(N,Kc)^T f16 + bias), f32 accumulate.
// 128 threads = 4 waves; each wave owns a 32x32 output tile (2x2 WMMA tiles,
// 4 v_wmma per K-step, each fragment reused twice). Rows are clamped (not
// guarded) so the K-loop is branch-free; guards only in the epilogue.
// Fragment layout: A and B both in contiguous k order, k = 16*hi + s.
// ---------------------------------------------------------------------------
__global__ void wmma_gemm_kernel(const half_t* __restrict__ A,
                                 const half_t* __restrict__ Wh,
                                 const float* __restrict__ bias,
                                 half_t* __restrict__ Ch, float* __restrict__ Cf,
                                 int M, int N, int Kc, int relu) {
  int lane = threadIdx.x & 31;
  int wv   = threadIdx.x >> 5;
  int lr = lane & 15;
  int hi = lane >> 4;                          // 0: lanes 0-15, 1: lanes 16-31
  int tm = (blockIdx.y * 4 + wv) * 32;
  int tn = blockIdx.x * 32;
  int m0 = tm + lr;      if (m0 > M - 1) m0 = M - 1;   // clamp: branch-free loop
  int m1 = tm + 16 + lr; if (m1 > M - 1) m1 = M - 1;
  const half_t* a0p = A + (size_t)m0 * Kc;
  const half_t* a1p = A + (size_t)m1 * Kc;
  const half_t* b0p = Wh + (size_t)(tn + lr) * Kc + (hi ? 16 : 0);
  const half_t* b1p = Wh + (size_t)(tn + 16 + lr) * Kc + (hi ? 16 : 0);
  v8f acc00 = {}, acc01 = {}, acc10 = {}, acc11 = {};
  for (int k0 = 0; k0 < Kc; k0 += 32) {
    ABfrag a0, a1, b0, b1;
    const v8h* ap0 = (const v8h*)(a0p + k0);
    const v8h* ap1 = (const v8h*)(a1p + k0);
    __builtin_prefetch(ap0 + 4, 0, 1);         // speculative; drops silently at end
    a0.h[0] = ap0[hi ? 2 : 0];                 // K 0-7  / 16-23
    a0.h[1] = ap0[hi ? 3 : 1];                 // K 8-15 / 24-31
    a1.h[0] = ap1[hi ? 2 : 0];
    a1.h[1] = ap1[hi ? 3 : 1];
    const v8h* bp0 = (const v8h*)(b0p + k0);
    const v8h* bp1 = (const v8h*)(b1p + k0);
    b0.h[0] = bp0[0]; b0.h[1] = bp0[1];        // K 0-15 / 16-31
    b1.h[0] = bp1[0]; b1.h[1] = bp1[1];
    acc00 = __builtin_amdgcn_wmma_f32_16x16x32_f16(false, a0.v, false, b0.v,
                                                   (short)0, acc00, false, false);
    acc01 = __builtin_amdgcn_wmma_f32_16x16x32_f16(false, a0.v, false, b1.v,
                                                   (short)0, acc01, false, false);
    acc10 = __builtin_amdgcn_wmma_f32_16x16x32_f16(false, a1.v, false, b0.v,
                                                   (short)0, acc10, false, false);
    acc11 = __builtin_amdgcn_wmma_f32_16x16x32_f16(false, a1.v, false, b1.v,
                                                   (short)0, acc11, false, false);
  }
  int n0 = tn + lr, n1 = tn + 16 + lr;
  float bv0 = bias ? bias[n0] : 0.f;
  float bv1 = bias ? bias[n1] : 0.f;
#pragma unroll
  for (int v = 0; v < 8; ++v) {
    int mm = tm + v + (hi ? 8 : 0);            // D layout: VGPR v <-> M = v + 8*hi
    if (mm < M) {
      float x0 = acc00[v] + bv0;
      float x1 = acc01[v] + bv1;
      if (relu) { x0 = fmaxf(x0, 0.f); x1 = fmaxf(x1, 0.f); }
      if (Ch) { Ch[(size_t)mm * N + n0] = (half_t)x0; Ch[(size_t)mm * N + n1] = (half_t)x1; }
      if (Cf) { Cf[(size_t)mm * N + n0] = x0;         Cf[(size_t)mm * N + n1] = x1; }
    }
    int mn = mm + 16;
    if (mn < M) {
      float x0 = acc10[v] + bv0;
      float x1 = acc11[v] + bv1;
      if (relu) { x0 = fmaxf(x0, 0.f); x1 = fmaxf(x1, 0.f); }
      if (Ch) { Ch[(size_t)mn * N + n0] = (half_t)x0; Ch[(size_t)mn * N + n1] = (half_t)x1; }
      if (Cf) { Cf[(size_t)mn * N + n0] = x0;         Cf[(size_t)mn * N + n1] = x1; }
    }
  }
}

// ---------------------------------------------------------------------------
// Max over the 32 samples of each center.
// ---------------------------------------------------------------------------
__global__ void maxpool_samples_kernel(const half_t* __restrict__ H,
                                       half_t* __restrict__ out, int npoint, int C) {
  int i = blockIdx.x * blockDim.x + threadIdx.x;
  if (i >= npoint * C) return;
  int s = i / C, c = i % C;
  float mx = -3.4e38f;
  for (int j = 0; j < 32; ++j)
    mx = fmaxf(mx, (float)H[((size_t)s * 32 + j) * C + c]);
  out[i] = (half_t)mx;
}

// Max over all centers of d (B,npts,C), packed into dst (B,dpitch) at doff.
__global__ void pool_centers_kernel(const half_t* __restrict__ d, int npts, int C,
                                    half_t* __restrict__ dst, int dpitch, int doff) {
  int c = blockIdx.x * blockDim.x + threadIdx.x;
  int b = blockIdx.y;
  if (c >= C) return;
  float mx = -3.4e38f;
  for (int s = 0; s < npts; ++s)
    mx = fmaxf(mx, (float)d[((size_t)b * npts + s) * C + c]);
  dst[(size_t)b * dpitch + doff + c] = (half_t)mx;
}

// agents_emb (32,256) -> mean over 8 agents -> f32 + f16 (4,256)
__global__ void agents_mean_kernel(const half_t* __restrict__ emb,
                                   float* __restrict__ af, half_t* __restrict__ ah) {
  int i = blockIdx.x * blockDim.x + threadIdx.x;
  if (i >= 4 * 256) return;
  int b = i / 256, c = i % 256;
  float s = 0.f;
  for (int a = 0; a < 8; ++a) s += (float)emb[((size_t)b * 8 + a) * 256 + c];
  s *= 0.125f;
  af[i] = s;
  ah[i] = (half_t)s;
}

// Single-query MHA: grid (heads=4, B=4), 64 threads (= keys = head_dim).
__global__ void attn_kernel(const float* __restrict__ qp, const half_t* __restrict__ kp,
                            const half_t* __restrict__ vp, half_t* __restrict__ o) {
  __shared__ float sc[64];
  int h = blockIdx.x, b = blockIdx.y, t = threadIdx.x;
  float s = 0.f;
  for (int d = 0; d < 64; ++d)
    s += qp[b * 256 + h * 64 + d] * (float)kp[((size_t)b * 64 + t) * 256 + h * 64 + d];
  sc[t] = s * 0.125f;                                  // 1/sqrt(64)
  __syncthreads();
  float mx = -3.4e38f;
  for (int i = 0; i < 64; ++i) mx = fmaxf(mx, sc[i]);
  float e = __expf(sc[t] - mx);
  __syncthreads();
  sc[t] = e;
  __syncthreads();
  float sum = 0.f;
  for (int i = 0; i < 64; ++i) sum += sc[i];
  float a = 0.f;
  for (int k = 0; k < 64; ++k)
    a += sc[k] * (float)vp[((size_t)b * 64 + k) * 256 + h * 64 + t];
  o[b * 256 + h * 64 + t] = (half_t)(a / sum);
}

// f32 (rows,cols) -> f16 packed into dst (rows,dpitch) at doff.
__global__ void pack_kernel(const float* __restrict__ src, half_t* __restrict__ dst,
                            int rows, int cols, int dpitch, int doff) {
  int i = blockIdx.x * blockDim.x + threadIdx.x;
  if (i >= rows * cols) return;
  int r = i / cols, c = i % cols;
  dst[(size_t)r * dpitch + doff + c] = (half_t)src[i];
}

// out[b] = h[b,:] . w1 + b1
__global__ void final_kernel(const half_t* __restrict__ h, const float* __restrict__ w1,
                             const float* __restrict__ b1, float* __restrict__ out) {
  int b = threadIdx.x;
  if (b >= 4) return;
  float s = b1[0];
  for (int c = 0; c < 256; ++c) s += (float)h[b * 256 + c] * w1[c];
  out[b] = s;
}

// ---------------------------------------------------------------------------
// Host orchestration
// ---------------------------------------------------------------------------
static inline int cdiv(int a, int b) { return (a + b - 1) / b; }

extern "C" void kernel_launch(void* const* d_in, const int* in_sizes, int n_in,
                              void* d_out, int out_size, void* d_ws, size_t ws_size,
                              hipStream_t stream) {
  (void)in_sizes; (void)n_in; (void)out_size; (void)ws_size;

  // jax tree_flatten (sorted keys) input order
  const float* cam        = (const float*)d_in[0];
  const float* env        = (const float*)d_in[1];
  const float* attn_in_b  = (const float*)d_in[2];
  const float* attn_in_w  = (const float*)d_in[3];
  const float* attn_out_b = (const float*)d_in[4];
  const float* attn_out_w = (const float*)d_in[5];
  const float* merge_b    = (const float*)d_in[6];
  const float* merge_w    = (const float*)d_in[7];
  const float* out_b0     = (const float*)d_in[8];
  const float* out_b1     = (const float*)d_in[9];
  const float* out_w0     = (const float*)d_in[10];
  const float* out_w1     = (const float*)d_in[11];
  const float* penc_b     = (const float*)d_in[12];
  const float* penc_w     = (const float*)d_in[13];

  size_t off = 0;
  auto alloc = [&](size_t bytes) -> void* {
    off = (off + 255) & ~(size_t)255;
    void* p = (char*)d_ws + off;
    off += bytes;
    return p;
  };
  auto gemm = [&](const half_t* A, const half_t* Wh, const float* bias,
                  half_t* Ch, float* Cf, int M, int N, int Kc, int relu) {
    dim3 g((unsigned)(N / 32), (unsigned)cdiv(M, 128));
    wmma_gemm_kernel<<<g, dim3(128), 0, stream>>>(A, Wh, bias, Ch, Cf, M, N, Kc, relu);
  };
  auto convert = [&](const float* W, int N, int Kw, int Kc) -> half_t* {
    half_t* Wh = (half_t*)alloc((size_t)N * Kc * sizeof(half_t));
    convert_w_kernel<<<cdiv(N * Kc, 256), 256, 0, stream>>>(W, Wh, N, Kw, Kc);
    return Wh;
  };

  // ---- network shape tables ----
  const int nL[5]     = {16384, 1024, 256, 64, 16};
  const float rad2[4] = {0.01f, 0.04f, 0.16f, 0.64f};
  const int Cin[4]    = {9, 64, 128, 256};
  const int cin3[4]   = {12, 67, 131, 259};
  const int kcin[4]   = {32, 96, 160, 288};
  const int chs[4][3] = {{32,32,64},{64,64,128},{128,128,256},{256,256,512}};

  // ---- convert all weights to padded f16 once per call ----
  half_t* sw[4][3];
  const float* sb[4][3];
  for (int l = 0; l < 4; ++l) {
    for (int j = 0; j < 3; ++j) sb[l][j] = (const float*)d_in[14 + 6 * l + j];
    sw[l][0] = convert((const float*)d_in[14 + 6 * l + 3], chs[l][0], cin3[l], kcin[l]);
    sw[l][1] = convert((const float*)d_in[14 + 6 * l + 4], chs[l][1], chs[l][0], chs[l][0]);
    sw[l][2] = convert((const float*)d_in[14 + 6 * l + 5], chs[l][2], chs[l][1], chs[l][1]);
  }
  half_t* penc_wh     = convert(penc_w, 256, 8, 32);
  half_t* merge_wh    = convert(merge_w, 64, 832, 832);
  half_t* attn_in_wh  = convert(attn_in_w, 768, 256, 256);
  half_t* attn_out_wh = convert(attn_out_w, 256, 256, 256);
  half_t* out0_wh     = convert(out_w0, 256, 576, 576);

  // ---- geometry (batch-shared) + per-batch features ----
  half_t* x_h    = (half_t*)alloc((size_t)4 * 16384 * 9 * sizeof(half_t));
  float* coordsL[5];
  coordsL[0] = (float*)alloc((size_t)16384 * 3 * sizeof(float));
  for (int l = 1; l < 5; ++l) coordsL[l] = (float*)alloc((size_t)nL[l] * 3 * sizeof(float));
  float* dist = (float*)alloc((size_t)16384 * sizeof(float));
  int* ball[4];
  for (int l = 0; l < 4; ++l) ball[l] = (int*)alloc((size_t)nL[l + 1] * 32 * sizeof(int));

  ply2data_kernel<<<cdiv(4 * 16384, 256), 256, 0, stream>>>(env, x_h, coordsL[0]);
  for (int l = 0; l < 4; ++l) {
    fps_kernel<<<1, 256, 0, stream>>>(coordsL[l], nL[l], nL[l + 1], coordsL[l + 1], dist);
    query_ball_kernel<<<cdiv(nL[l + 1], QB_BLK), QB_BLK, 0, stream>>>(
        coordsL[l], nL[l], coordsL[l + 1], nL[l + 1], rad2[l], ball[l]);
  }

  // ---- SA layer MLP chains (WMMA), per batch ----
  half_t* dbuf[5];
  dbuf[0] = x_h;
  for (int l = 0; l < 4; ++l)
    dbuf[l + 1] = (half_t*)alloc((size_t)4 * nL[l + 1] * chs[l][2] * sizeof(half_t));
  half_t* Asc = (half_t*)alloc((size_t)1048576 * sizeof(half_t));   // max M*Kc
  half_t* H1  = (half_t*)alloc((size_t)2097152 * sizeof(half_t));   // max M*Cout
  half_t* H2  = (half_t*)alloc((size_t)2097152 * sizeof(half_t));

  for (int l = 0; l < 4; ++l) {
    int np = nL[l + 1], M = np * 32;
    for (int b = 0; b < 4; ++b) {
      const half_t* feats = dbuf[l] + (size_t)b * nL[l] * Cin[l];
      group_kernel<<<cdiv(M, 256), 256, 0, stream>>>(
          feats, Cin[l], coordsL[l], coordsL[l + 1], ball[l], Asc, np, kcin[l]);
      gemm(Asc, sw[l][0], sb[l][0], H1, nullptr, M, chs[l][0], kcin[l], 1);
      gemm(H1,  sw[l][1], sb[l][1], H2, nullptr, M, chs[l][1], chs[l][0], 1);
      gemm(H2,  sw[l][2], sb[l][2], H1, nullptr, M, chs[l][2], chs[l][1], 1);
      maxpool_samples_kernel<<<cdiv(np * chs[l][2], 256), 256, 0, stream>>>(
          H1, dbuf[l + 1] + (size_t)b * np * chs[l][2], np, chs[l][2]);
    }
  }

  // ---- global max pools -> mr (B,832) f16 ----
  half_t* mr_h = (half_t*)alloc((size_t)4 * 832 * sizeof(half_t));
  pool_centers_kernel<<<dim3(cdiv(64, 64), 4),  64, 0, stream>>>(dbuf[1], 1024, 64,  mr_h, 832, 0);
  pool_centers_kernel<<<dim3(cdiv(256, 64), 4), 64, 0, stream>>>(dbuf[3], 64,  256, mr_h, 832, 64);
  pool_centers_kernel<<<dim3(cdiv(512, 64), 4), 64, 0, stream>>>(dbuf[4], 16,  512, mr_h, 832, 320);

  // ---- heads ----
  half_t* cam_h = convert(cam, 32, 8, 32);                          // (B*8, 8) padded
  half_t* emb_h = (half_t*)alloc((size_t)32 * 256 * sizeof(half_t));
  gemm(cam_h, penc_wh, penc_b, emb_h, nullptr, 32, 256, 32, 1);
  float*  agents_f = (float*)alloc((size_t)4 * 256 * sizeof(float));
  half_t* agents_h = (half_t*)alloc((size_t)4 * 256 * sizeof(half_t));
  agents_mean_kernel<<<cdiv(1024, 256), 256, 0, stream>>>(emb_h, agents_f, agents_h);

  float* envf = (float*)alloc((size_t)4 * 64 * sizeof(float));
  gemm(mr_h, merge_wh, merge_b, nullptr, envf, 4, 64, 832, 1);

  float*  qp = (float*)alloc((size_t)4 * 256 * sizeof(float));
  half_t* kp = (half_t*)alloc((size_t)256 * 256 * sizeof(half_t));
  half_t* vp = (half_t*)alloc((size_t)256 * 256 * sizeof(half_t));
  gemm(agents_h, attn_in_wh,             attn_in_b,       nullptr, qp, 4,   256, 256, 0);
  gemm(dbuf[3],  attn_in_wh + 256 * 256, attn_in_b + 256, kp, nullptr, 256, 256, 256, 0);
  gemm(dbuf[3],  attn_in_wh + 512 * 256, attn_in_b + 512, vp, nullptr, 256, 256, 256, 0);

  half_t* o_h = (half_t*)alloc((size_t)4 * 256 * sizeof(half_t));
  attn_kernel<<<dim3(4, 4), 64, 0, stream>>>(qp, kp, vp, o_h);
  float* attnf = (float*)alloc((size_t)4 * 256 * sizeof(float));
  gemm(o_h, attn_out_wh, attn_out_b, nullptr, attnf, 4, 256, 256, 0);

  // merged = [env_feats(64) | agents(256) | attn(256)] -> (B,576) f16
  half_t* merged = (half_t*)alloc((size_t)4 * 576 * sizeof(half_t));
  pack_kernel<<<cdiv(4 * 64, 256),  256, 0, stream>>>(envf,     merged, 4, 64,  576, 0);
  pack_kernel<<<cdiv(4 * 256, 256), 256, 0, stream>>>(agents_f, merged, 4, 256, 576, 64);
  pack_kernel<<<cdiv(4 * 256, 256), 256, 0, stream>>>(attnf,    merged, 4, 256, 576, 320);

  half_t* hb = (half_t*)alloc((size_t)4 * 256 * sizeof(half_t));
  gemm(merged, out0_wh, out_b0, hb, nullptr, 4, 256, 576, 1);
  final_kernel<<<1, 32, 0, stream>>>(hb, out_w1, out_b1, (float*)d_out);
}
